// SecondOrderRelationalAttention_28656021799446
// MI455X (gfx1250) — matrix operand order repaired and checked by reading, not compiled, not run on hardware
//
#include <hip/hip_runtime.h>
#include <math.h>

// Problem constants
#define BB 2
#define CC 512
#define DD 512
#define HH 8
#define DH 64

typedef __attribute__((ext_vector_type(16))) __bf16 v16bf;
typedef __attribute__((ext_vector_type(8)))  float  v8f;

#define DEV __device__ __forceinline__

DEV v8f zero8() {
    v8f z;
#pragma unroll
    for (int r = 0; r < 8; ++r) z[r] = 0.0f;
    return z;
}
DEV v8f splat8(float x) {
    v8f z;
#pragma unroll
    for (int r = 0; r < 8; ++r) z[r] = x;
    return z;
}

union frag_cast { uint4 u[2]; v16bf v; };

// A-matrix fragment: 16x32 bf16, row-major in LDS (ld elems/row, ld%8==0).
// lane L<16 -> row L, K 0..15; lane>=16 -> row L-16, K 16..31 (matches B fragments).
// Each lane's 16 values are two contiguous 16-byte runs -> 2x ds_load_b128.
DEV v16bf load_A_frag(const __bf16* A, int ld) {
    const int lane = threadIdx.x & 31;
    const int half = lane >> 4;
    const int row  = lane & 15;
    frag_cast t;
    t.u[0] = *(const uint4*)(A + row * ld + 16 * half);
    t.u[1] = *(const uint4*)(A + row * ld + 16 * half + 8);
    return t.v;
}

// B-matrix fragment from a COLUMN-MAJOR tile BT[col*ldk + k] (ldk%8==0).
// lanes 0-15: N=lane, K=0..15 ; lanes 16-31: N=lane-16, K=16..31.
// Per lane: 16 contiguous k values -> 2x ds_load_b128.
DEV v16bf load_BT_frag(const __bf16* BT, int ldk) {
    const int lane = threadIdx.x & 31;
    const int half = lane >> 4;
    const int col  = lane & 15;
    frag_cast t;
    const uint4* p = (const uint4*)(BT + col * ldk + 16 * half);
    t.u[0] = p[0];
    t.u[1] = p[1];
    return t.v;
}

// B-matrix fragment, row-major 32x16 source (gather path, used once for attn@V).
DEV v16bf load_B_rm(const __bf16* Bm, int ld) {
    const int lane = threadIdx.x & 31;
    const int half = lane >> 4;
    const int col  = lane & 15;
    v16bf b;
#pragma unroll
    for (int p = 0; p < 16; ++p) {
        const int pv = p >> 1;
        const int kk = 2 * pv + (p & 1) + 16 * half;
        b[p] = Bm[kk * ld + col];
    }
    return b;
}

DEV v8f wmma_bf16(v16bf a, v16bf b, v8f c) {
    return __builtin_amdgcn_wmma_f32_16x16x32_bf16(false, a, false, b, (short)0, c, false, false);
}

DEV float gelu_exact(float x) {
    return 0.5f * x * (1.0f + erff(x * 0.70710678118654752f));
}

// gfx1250 async global->LDS copy (16B per lane), tracked by ASYNCcnt.
DEV void async_g2l_b128(unsigned lds_byte_off, const uint4* g) {
    asm volatile("global_load_async_to_lds_b128 %0, %1, off"
                 :: "v"(lds_byte_off), "v"(g)
                 : "memory");
}
DEV void wait_async0() { asm volatile("s_wait_asynccnt 0" ::: "memory"); }

// ---------------------------------------------------------------------------
// Kernel 1: QKV projection. x(1024x512) @ W(512x512) + bias -> bf16 (B,H,C,DH)
// grid (16, 8, 3), block 128 (4 waves). 64x64 output tile per workgroup.
// ---------------------------------------------------------------------------
__global__ __launch_bounds__(128) void k_qkv_proj(
    const float* __restrict__ x,
    const float* __restrict__ Wq, const float* __restrict__ bq,
    const float* __restrict__ Wk, const float* __restrict__ bk,
    const float* __restrict__ Wv, const float* __restrict__ bv,
    __bf16* __restrict__ Qb, __bf16* __restrict__ Kb, __bf16* __restrict__ Vb)
{
    const int z = blockIdx.z;
    const float* W    = (z == 0) ? Wq : (z == 1 ? Wk : Wv);
    const float* bias = (z == 0) ? bq : (z == 1 ? bk : bv);
    __bf16* out       = (z == 0) ? Qb : (z == 1 ? Kb : Vb);

    __shared__ __bf16 As[64 * 32];       // row-major
    __shared__ __bf16 BsT[64 * 32];      // column-major: BsT[c*32 + r]

    const int tid = threadIdx.x, lane = tid & 31, wv = tid >> 5;
    const int half = lane >> 4, nl = lane & 15;
    const int mt = blockIdx.x;   // row tile (64 rows), 0..15
    const int nb = blockIdx.y;   // col tile (64 cols), 0..7

    v8f acc[4];
#pragma unroll
    for (int nt = 0; nt < 4; ++nt) acc[nt] = zero8();

    for (int ks = 0; ks < 16; ++ks) {
#pragma unroll
        for (int u = 0; u < 16; ++u) {
            const int idx = tid * 16 + u;       // 0..2047
            const int r = idx >> 5, c = idx & 31;
            As[idx] = (__bf16)x[(mt * 64 + r) * DD + ks * 32 + c];
        }
#pragma unroll
        for (int u = 0; u < 16; ++u) {
            const int idx = tid * 16 + u;       // 0..2047
            const int r = idx >> 6, c = idx & 63;
            BsT[c * 32 + r] = (__bf16)W[(ks * 32 + r) * DD + nb * 64 + c];
        }
        if (ks + 1 < 16)
            __builtin_prefetch(&x[(mt * 64 + (tid >> 1)) * DD + (ks + 1) * 32], 0, 1);
        __syncthreads();

        const v16bf a = load_A_frag(As + (wv * 16) * 32, 32);
#pragma unroll
        for (int nt = 0; nt < 4; ++nt) {
            const v16bf b = load_BT_frag(BsT + (nt * 16) * 32, 32);
            acc[nt] = wmma_bf16(a, b, acc[nt]);
        }
        __syncthreads();
    }

#pragma unroll
    for (int nt = 0; nt < 4; ++nt) {
#pragma unroll
        for (int r = 0; r < 8; ++r) {
            const int m = mt * 64 + wv * 16 + r + 8 * half;   // 0..1023
            const int n = nb * 64 + nt * 16 + nl;             // 0..511
            const float v = acc[nt][r] + bias[n];
            const int b = m >> 9, c = m & 511, h = n >> 6, dh = n & 63;
            out[(((b * HH + h) * CC) + c) * DH + dh] = (__bf16)v;
        }
    }
}

// ---------------------------------------------------------------------------
// Kernel 2: hq = q@W1q + b1 (f32 out, (B,H,C,64)).  grid (16, 8), block 128.
// ---------------------------------------------------------------------------
__global__ __launch_bounds__(128) void k_headproj(
    const __bf16* __restrict__ Qb,
    const float* __restrict__ W1, const float* __restrict__ b1,
    float* __restrict__ hq)
{
    const int bh = blockIdx.x;                // 0..15
    const int ct = blockIdx.y;                // 0..7 (64 rows)

    __shared__ __bf16 As[64 * 64];            // row-major q tile
    __shared__ __bf16 BT[64 * 64];            // column-major W1q: BT[e*64 + d]

    const int tid = threadIdx.x, lane = tid & 31, wv = tid >> 5;
    const int half = lane >> 4, nl = lane & 15;

#pragma unroll
    for (int u = 0; u < 32; ++u) {
        const int idx = tid + u * 128;        // 0..4095
        As[idx] = Qb[(bh * CC + ct * 64) * DH + idx];
        const int d = idx >> 6, e = idx & 63;
        BT[e * 64 + d] = (__bf16)W1[idx];     // W1q = W1[0:64]
    }
    __syncthreads();

    v8f acc[4];
#pragma unroll
    for (int nt = 0; nt < 4; ++nt) acc[nt] = zero8();
#pragma unroll
    for (int ksp = 0; ksp < 2; ++ksp) {
        const v16bf a = load_A_frag(As + (wv * 16) * 64 + ksp * 32, 64);
#pragma unroll
        for (int nt = 0; nt < 4; ++nt) {
            const v16bf b = load_BT_frag(BT + (nt * 16) * 64 + ksp * 32, 64);
            acc[nt] = wmma_bf16(a, b, acc[nt]);
        }
    }

#pragma unroll
    for (int nt = 0; nt < 4; ++nt) {
#pragma unroll
        for (int r = 0; r < 8; ++r) {
            const int row = ct * 64 + wv * 16 + r + 8 * half;
            const int e   = nt * 16 + nl;
            hq[(bh * CC + row) * DH + e] = acc[nt][r] + b1[e];
        }
    }
}

// ---------------------------------------------------------------------------
// Kernel 3: pairwise scores + softmax + attn@V. One workgroup = (b,h, 16 i's).
// h[i,j,e] = sum_d k[j,d]*(q_i[d]*W1i[d,e] + W1k[d,e]) + hq[i,e]  (hk fused!)
// grid (32, 16), block 256 (8 waves), dynamic LDS ~142 KB.
// ---------------------------------------------------------------------------
#define OFF_KS    0u        // 512*64 bf16 = 65536  (K tile, later V tile)
#define OFF_ATTN  65536u    // 16*512 bf16 = 16384
#define OFF_SC    81920u    // 16*512 f32  = 32768
#define OFF_MIT   114688u   // 64*64 bf16  = 8192   (M_i' column-major [e][d])
#define OFF_W1IT  122880u   // 64*64 bf16  = 8192   (W1i^T [e][d])
#define OFF_W1KT  131072u   // 64*64 bf16  = 8192   (W1k^T [e][d])
#define OFF_QS    139264u   // 16*64 bf16  = 2048
#define OFF_HQS   141312u   // 16*64 f32   = 4096
#define OFF_W2    145408u   // 64 f32      = 256
#define PW_LDS_BYTES 145664u

__global__ __launch_bounds__(256) void k_pairwise(
    const __bf16* __restrict__ Qb, const __bf16* __restrict__ Kb,
    const __bf16* __restrict__ Vb,
    const float* __restrict__ hq,
    const float* __restrict__ W1, const float* __restrict__ W2,
    const float* __restrict__ b2,
    __bf16* __restrict__ Ob)
{
    extern __shared__ __align__(16) char smem_raw[];
    __bf16* Ks   = (__bf16*)(smem_raw + OFF_KS);
    __bf16* attn = (__bf16*)(smem_raw + OFF_ATTN);
    float*  sc_s = (float*) (smem_raw + OFF_SC);
    __bf16* MiT  = (__bf16*)(smem_raw + OFF_MIT);
    __bf16* W1iT = (__bf16*)(smem_raw + OFF_W1IT);
    __bf16* W1kT = (__bf16*)(smem_raw + OFF_W1KT);
    __bf16* q_s  = (__bf16*)(smem_raw + OFF_QS);
    float*  hq_s = (float*) (smem_raw + OFF_HQS);
    float*  w2_s = (float*) (smem_raw + OFF_W2);

    const int tid = threadIdx.x, lane = tid & 31, wv = tid >> 5;
    const int half = lane >> 4, nl = lane & 15;
    const int itile = blockIdx.x;       // 0..31
    const int bh    = blockIdx.y;       // 0..15
    const int hd    = bh & (HH - 1);
    const int bb    = bh >> 3;

    // --- K tile via async DMA to LDS; small tiles via normal loads ---
    {
        const uint4* Kg4 = (const uint4*)(Kb + (long)bh * CC * DH);
#pragma unroll
        for (int u = 0; u < 16; ++u) {
            const int vec = tid + u * 256;
            async_g2l_b128(OFF_KS + vec * 16u, Kg4 + vec);
        }
        const __bf16* qg = Qb + ((long)bh * CC + itile * 16) * DH;
        const float*  hg = hq + ((long)bh * CC + itile * 16) * DH;
#pragma unroll
        for (int u = 0; u < 4; ++u) {
            const int idx = tid + u * 256;
            q_s[idx]  = qg[idx];
            hq_s[idx] = hg[idx];
        }
        const float* w1kg = W1 + DH * DH;       // W1k = W1[64:128]
        const float* w1ig = W1 + 2 * DH * DH;   // W1i = W1[128:192]
#pragma unroll
        for (int u = 0; u < 16; ++u) {
            const int idx = tid + u * 256;
            const int d = idx >> 6, e = idx & 63;
            W1iT[e * 64 + d] = (__bf16)w1ig[idx];
            W1kT[e * 64 + d] = (__bf16)w1kg[idx];
        }
        if (tid < 64) w2_s[tid] = W2[tid];
    }
    const float b2v = b2[0];
    wait_async0();
    __syncthreads();

    // A-fragments of K for this wave's 4 j-tiles (i-independent): keep in regs
    v16bf afr[4][2];
#pragma unroll
    for (int jj = 0; jj < 4; ++jj) {
        const int jt = wv * 4 + jj;
#pragma unroll
        for (int ksp = 0; ksp < 2; ++ksp)
            afr[jj][ksp] = load_A_frag(Ks + (jt * 16) * DH + ksp * 32, DH);
    }

    // --- per-i fused GEMM + gelu + score ---
    for (int i = 0; i < 16; ++i) {
        __syncthreads();    // everyone finished reading previous MiT
        // MiT[e][d] = q_i[d]*W1i[d][e] + W1k[d][e]   (hk folded into the GEMM)
#pragma unroll
        for (int u = 0; u < 16; ++u) {
            const int idx = tid + u * 256;           // = e*64 + d
            const int d = idx & 63;
            MiT[idx] = (__bf16)((float)q_s[i * DH + d] * (float)W1iT[idx]
                                + (float)W1kT[idx]);
        }
        __syncthreads();

        v16bf bfr[4][2];
#pragma unroll
        for (int et = 0; et < 4; ++et)
#pragma unroll
            for (int ksp = 0; ksp < 2; ++ksp)
                bfr[et][ksp] = load_BT_frag(MiT + (et * 16) * DH + ksp * 32, DH);

#pragma unroll
        for (int jj = 0; jj < 4; ++jj) {
            const int jt16 = (wv * 4 + jj) * 16;
            float sp[8];
#pragma unroll
            for (int r = 0; r < 8; ++r) sp[r] = 0.0f;

#pragma unroll
            for (int et = 0; et < 4; ++et) {
                const int e = et * 16 + nl;
                const float w2v = w2_s[e];
                v8f acc = splat8(hq_s[i * DH + e]);   // hq (+b1) as C init
                acc = wmma_bf16(afr[jj][0], bfr[et][0], acc);
                acc = wmma_bf16(afr[jj][1], bfr[et][1], acc);
#pragma unroll
                for (int r = 0; r < 8; ++r)
                    sp[r] += gelu_exact(acc[r]) * w2v;
            }
#pragma unroll
            for (int r = 0; r < 8; ++r) {
                float v = sp[r];
                v += __shfl_xor(v, 1, 32);
                v += __shfl_xor(v, 2, 32);
                v += __shfl_xor(v, 4, 32);
                v += __shfl_xor(v, 8, 32);
                if (nl == 0)
                    sc_s[i * CC + jt16 + r + 8 * half] = (v + b2v) * 0.125f;
            }
        }
    }
    __syncthreads();

    // kick off V tile DMA into the K region; overlaps with softmax below
    {
        const uint4* Vg4 = (const uint4*)(Vb + (long)bh * CC * DH);
#pragma unroll
        for (int u = 0; u < 16; ++u) {
            const int vec = tid + u * 256;
            async_g2l_b128(OFF_KS + vec * 16u, Vg4 + vec);
        }
    }

    // softmax over j (512) per i row -> attn (bf16)
#pragma unroll
    for (int rr = 0; rr < 2; ++rr) {
        const int row = wv * 2 + rr;
        float mx = -1e30f;
        for (int t = lane; t < CC; t += 32) mx = fmaxf(mx, sc_s[row * CC + t]);
        mx = fmaxf(mx, __shfl_xor(mx, 16, 32));
        mx = fmaxf(mx, __shfl_xor(mx, 8, 32));
        mx = fmaxf(mx, __shfl_xor(mx, 4, 32));
        mx = fmaxf(mx, __shfl_xor(mx, 2, 32));
        mx = fmaxf(mx, __shfl_xor(mx, 1, 32));
        float sum = 0.0f;
        for (int t = lane; t < CC; t += 32) {
            const float ev = __expf(sc_s[row * CC + t] - mx);
            sum += ev;
            sc_s[row * CC + t] = ev;
        }
        sum += __shfl_xor(sum, 16, 32);
        sum += __shfl_xor(sum, 8, 32);
        sum += __shfl_xor(sum, 4, 32);
        sum += __shfl_xor(sum, 2, 32);
        sum += __shfl_xor(sum, 1, 32);
        const float inv = 1.0f / sum;
        for (int t = lane; t < CC; t += 32)
            attn[row * CC + t] = (__bf16)(sc_s[row * CC + t] * inv);
    }
    wait_async0();
    __syncthreads();

    // attn (16x512) @ V (512x64): waves 0..3 take one 16-col d-tile each
    if (wv < 4) {
        v8f acc = zero8();
#pragma unroll
        for (int ksp = 0; ksp < 16; ++ksp) {
            const v16bf a = load_A_frag(attn + ksp * 32, CC);
            const v16bf b = load_B_rm(Ks + (ksp * 32) * DH + wv * 16, DH);
            acc = wmma_bf16(a, b, acc);
        }
#pragma unroll
        for (int r = 0; r < 8; ++r) {
            const int c = itile * 16 + r + 8 * half;
            const int d = wv * 16 + nl;
            Ob[((long)(bb * CC + c)) * DD + hd * DH + d] = (__bf16)acc[r];
        }
    }
}

// ---------------------------------------------------------------------------
// Kernel 4: output projection. Ob(1024x512 bf16) @ Wo + bo -> f32 out (B,C,D)
// ---------------------------------------------------------------------------
__global__ __launch_bounds__(128) void k_out_proj(
    const __bf16* __restrict__ Ob, const float* __restrict__ Wo,
    const float* __restrict__ bo, float* __restrict__ Y)
{
    __shared__ __bf16 As[64 * 32];
    __shared__ __bf16 BsT[64 * 32];

    const int tid = threadIdx.x, lane = tid & 31, wv = tid >> 5;
    const int half = lane >> 4, nl = lane & 15;
    const int mt = blockIdx.x, nb = blockIdx.y;

    v8f acc[4];
#pragma unroll
    for (int nt = 0; nt < 4; ++nt) acc[nt] = zero8();

    for (int ks = 0; ks < 16; ++ks) {
#pragma unroll
        for (int u = 0; u < 16; ++u) {
            const int idx = tid * 16 + u;
            const int r = idx >> 5, c = idx & 31;
            As[idx] = Ob[(mt * 64 + r) * DD + ks * 32 + c];
        }
#pragma unroll
        for (int u = 0; u < 16; ++u) {
            const int idx = tid * 16 + u;
            const int r = idx >> 6, c = idx & 63;
            BsT[c * 32 + r] = (__bf16)Wo[(ks * 32 + r) * DD + nb * 64 + c];
        }
        __syncthreads();

        const v16bf a = load_A_frag(As + (wv * 16) * 32, 32);
#pragma unroll
        for (int nt = 0; nt < 4; ++nt) {
            const v16bf b = load_BT_frag(BsT + (nt * 16) * 32, 32);
            acc[nt] = wmma_bf16(a, b, acc[nt]);
        }
        __syncthreads();
    }

#pragma unroll
    for (int nt = 0; nt < 4; ++nt) {
#pragma unroll
        for (int r = 0; r < 8; ++r) {
            const int m = mt * 64 + wv * 16 + r + 8 * half;
            const int n = nb * 64 + nt * 16 + nl;
            Y[(long)m * DD + n] = acc[nt][r] + bo[n];
        }
    }
}

// ---------------------------------------------------------------------------
extern "C" void kernel_launch(void* const* d_in, const int* in_sizes, int n_in,
                              void* d_out, int out_size, void* d_ws, size_t ws_size,
                              hipStream_t stream) {
    const float* x  = (const float*)d_in[0];
    const float* Wq = (const float*)d_in[1];
    const float* bq = (const float*)d_in[2];
    const float* Wk = (const float*)d_in[3];
    const float* bk = (const float*)d_in[4];
    const float* Wv = (const float*)d_in[5];
    const float* bv = (const float*)d_in[6];
    const float* W1 = (const float*)d_in[7];
    const float* b1 = (const float*)d_in[8];
    const float* W2 = (const float*)d_in[9];
    const float* b2 = (const float*)d_in[10];
    const float* Wo = (const float*)d_in[11];
    const float* bo = (const float*)d_in[12];
    float* out = (float*)d_out;

    char* ws = (char*)d_ws;
    const size_t MB = 1 << 20;
    __bf16* Qb = (__bf16*)(ws + 0 * MB);   // B*H*C*DH bf16 = 1 MB
    __bf16* Kb = (__bf16*)(ws + 1 * MB);
    __bf16* Vb = (__bf16*)(ws + 2 * MB);
    float*  hq = (float*) (ws + 3 * MB);   // 2 MB
    __bf16* Ob = (__bf16*)(ws + 5 * MB);   // 1 MB

    (void)hipFuncSetAttribute(reinterpret_cast<const void*>(k_pairwise),
                              hipFuncAttributeMaxDynamicSharedMemorySize,
                              (int)PW_LDS_BYTES);

    k_qkv_proj<<<dim3(16, 8, 3), 128, 0, stream>>>(x, Wq, bq, Wk, bk, Wv, bv, Qb, Kb, Vb);
    k_headproj<<<dim3(16, 8), 128, 0, stream>>>(Qb, W1, b1, hq);
    k_pairwise<<<dim3(32, 16), 256, PW_LDS_BYTES, stream>>>(Qb, Kb, Vb, hq, W1, W2, b2, Ob);
    k_out_proj<<<dim3(16, 8, 1), 128, 0, stream>>>(Ob, Wo, bo, out);
}
